// CausalAttentionProduct_6485400617275
// MI455X (gfx1250) — matrix operand order repaired and checked by reading, not compiled, not run on hardware
//
#include <hip/hip_runtime.h>

typedef __attribute__((ext_vector_type(16))) __bf16 v16bf;
typedef __attribute__((ext_vector_type(8)))  __bf16 v8bf;
typedef __attribute__((ext_vector_type(8)))  float  v8f;

#define S_LEN  4096
#define DHEAD  64
#define NHEAD  12
#define NBH    24          // B*H
#define QTILE  128         // queries per workgroup (8 waves x 16)
#define KTILE  64          // keys per iteration
#define NITER  (S_LEN / KTILE)   // 64
#define NQT    (S_LEN / QTILE)   // 32
#define KPITCH 72          // bf16 elements per sK row (64 + pad)
#define VPITCH 72          // bf16 elements per sV row (64 keys + pad)
#define OPITCH 68          // f32 elements per sO row (64 + pad)
#define FMIN   (-3.4028234663852886e38f)
#define LOG2E  1.44269504088896340736f
#define QSCALE (0.125f * LOG2E)   // 1/sqrt(64) * log2(e): softmax in base-2 domain

#define SK_BYTES  (KTILE * KPITCH * 2)     // 9216
#define SV_BYTES  (DHEAD * VPITCH * 2)     // 9216
#define SKV_BYTES (SK_BYTES + SV_BYTES)    // 18432 per buffer (x2)
#define SO_BYTES  (8 * 16 * OPITCH * 4)    // 34816
#define SMEM_BYTES (2 * SKV_BYTES)         // 36864 >= SO_BYTES

__device__ __forceinline__ v16bf cat16(v8bf lo, v8bf hi) {
  return __builtin_shufflevector(lo, hi, 0,1,2,3,4,5,6,7,8,9,10,11,12,13,14,15);
}

static __device__ __forceinline__ unsigned pack2bf(float a, float b) {
  union { __bf16 h[2]; unsigned u; } x;
  x.h[0] = (__bf16)a; x.h[1] = (__bf16)b;
  return x.u;
}

union B8 { v8bf v; unsigned u[4]; };
union PB { v16bf v; unsigned u[8]; };

// two float4 (8 consecutive floats) -> v8bf via clean v_cvt_pk_bf16_f32 pairs
static __device__ __forceinline__ v8bf cvt8(float4 a, float4 b) {
  B8 x;
  x.u[0] = pack2bf(a.x, a.y);
  x.u[1] = pack2bf(a.z, a.w);
  x.u[2] = pack2bf(b.x, b.y);
  x.u[3] = pack2bf(b.z, b.w);
  return x.v;
}

struct Stage {
  float4 k0a, k0b, k1a, k1b;   // K rows rK and rK+32, 8 floats each
  float4 v0a, v0b, v1a, v1b;   // V key rows 2u and 2u+1, 8 floats each
};

__global__ __launch_bounds__(256, 2)
void fa_bf16_wmma_t_kernel(const float* __restrict__ Q, const float* __restrict__ K,
                           const float* __restrict__ V, const float* __restrict__ M,
                           float* __restrict__ O)
{
  __shared__ __align__(16) char smem[SMEM_BYTES];

  const int bh = blockIdx.x / NQT;
  const int qt = blockIdx.x % NQT;
  const int b  = bh / NHEAD;

  const float* qg = Q + (size_t)bh * S_LEN * DHEAD;
  const float* kg = K + (size_t)bh * S_LEN * DHEAD;
  const float* vg = V + (size_t)bh * S_LEN * DHEAD;
  const float* mg = M + (size_t)b  * S_LEN;
  float*       og = O + (size_t)bh * S_LEN * DHEAD;

  const int tid  = threadIdx.x;
  const int wave = tid >> 5;
  const int lane = tid & 31;
  const int ln   = lane & 15;
  const bool hi  = lane >= 16;
  const int  hb  = hi ? 8 : 0;     // key/d sub-offset for this lane half (C layouts; A loads use 2*hb)

  const int q0 = qt * QTILE + wave * 16;

  // staging thread mappings (256 threads stage 64x64 K and 64x64 V per tile)
  const int rK = tid >> 3;          // K: key rows rK and rK+32
  const int cK = (tid & 7) * 8;     // K: d offset 0..56
  const int uV = tid >> 3;          // V: key pair -> keys {2u, 2u+1}
  const int wV = (tid & 7) * 8;     // V: d offsets wV..wV+7

  // ---- Load Q^T in WMMA B-layout (lane = query column, d packed), scale folded ----
  v16bf qb[2];
  {
    const int dof = hi ? 16 : 0;
    #pragma unroll
    for (int c = 0; c < 2; ++c) {
      const float* qp = qg + (size_t)(q0 + ln) * DHEAD + c * 32 + dof;
      float4 x0 = *(const float4*)(qp);
      float4 x1 = *(const float4*)(qp + 4);
      float4 x2 = *(const float4*)(qp + 8);
      float4 x3 = *(const float4*)(qp + 12);
      float f[16] = {x0.x,x0.y,x0.z,x0.w, x1.x,x1.y,x1.z,x1.w,
                     x2.x,x2.y,x2.z,x2.w, x3.x,x3.y,x3.z,x3.w};
      v16bf t;
      #pragma unroll
      for (int j = 0; j < 16; ++j) t[j] = (__bf16)(f[j] * QSCALE);
      qb[c] = t;
    }
  }

  v8f acc[4];
  #pragma unroll
  for (int n = 0; n < 4; ++n)
    #pragma unroll
    for (int j = 0; j < 8; ++j) acc[n][j] = 0.0f;
  float m_state = -__builtin_inff();
  float l_state = 0.0f;

  // ---- staging helpers (lambdas keep mappings in scope) ----
  auto stage_load = [&](int key0) -> Stage {
    Stage s;
    const float* kp0 = kg + (size_t)(key0 + rK) * DHEAD + cK;
    const float* kp1 = kp0 + (size_t)32 * DHEAD;
    s.k0a = *(const float4*)kp0; s.k0b = *(const float4*)(kp0 + 4);
    s.k1a = *(const float4*)kp1; s.k1b = *(const float4*)(kp1 + 4);
    const float* vp0 = vg + (size_t)(key0 + 2 * uV) * DHEAD + wV;
    const float* vp1 = vp0 + DHEAD;
    s.v0a = *(const float4*)vp0; s.v0b = *(const float4*)(vp0 + 4);
    s.v1a = *(const float4*)vp1; s.v1b = *(const float4*)(vp1 + 4);
    return s;
  };
  auto stage_store = [&](char* buf, const Stage& s) {
    __bf16* nK = (__bf16*)buf;
    *(v8bf*)&nK[rK * KPITCH + cK]        = cvt8(s.k0a, s.k0b);
    *(v8bf*)&nK[(rK + 32) * KPITCH + cK] = cvt8(s.k1a, s.k1b);
    __bf16* nV = (__bf16*)(buf + SK_BYTES);
    float fa[8] = {s.v0a.x,s.v0a.y,s.v0a.z,s.v0a.w, s.v0b.x,s.v0b.y,s.v0b.z,s.v0b.w};
    float fb[8] = {s.v1a.x,s.v1a.y,s.v1a.z,s.v1a.w, s.v1b.x,s.v1b.y,s.v1b.z,s.v1b.w};
    #pragma unroll
    for (int j = 0; j < 8; ++j)
      *(unsigned*)&nV[(wV + j) * VPITCH + 2 * uV] = pack2bf(fa[j], fb[j]);
  };

  // ---- prologue: stage tile 0 into buffer 0 ----
  {
    Stage s0 = stage_load(0);
    stage_store(smem, s0);
  }
  __syncthreads();

  for (int kb = 0; kb < NITER; ++kb) {
    const int key0 = kb * KTILE;
    char* curb = smem + (size_t)(kb & 1) * SKV_BYTES;
    char* nxtb = smem + (size_t)((kb + 1) & 1) * SKV_BYTES;
    __bf16* sK = (__bf16*)curb;
    __bf16* sV = (__bf16*)(curb + SK_BYTES);

    // issue next tile's global loads early (last iter: dead reload of tile 0)
    const int nkey0 = (kb + 1 < NITER) ? key0 + KTILE : 0;
    Stage ns = stage_load(nkey0);

    // ---- S^T = K * Q^T : 4 key tiles of 16, D split as 2 x K32 -> 8 WMMA ----
    v8f st[4];
    {
      v8f z;
      #pragma unroll
      for (int j = 0; j < 8; ++j) z[j] = 0.0f;
      #pragma unroll
      for (int t = 0; t < 4; ++t) {
        const int krow = (t * 16 + ln) * KPITCH;
        v16bf a0 = cat16(*(v8bf*)&sK[krow +      2 * hb],
                         *(v8bf*)&sK[krow +      2 * hb + 8]);
        v16bf a1 = cat16(*(v8bf*)&sK[krow + 32 + 2 * hb],
                         *(v8bf*)&sK[krow + 32 + 2 * hb + 8]);
        st[t] = __builtin_amdgcn_wmma_f32_16x16x32_bf16(false, a0, false, qb[0], (short)0, z,     false, false);
        st[t] = __builtin_amdgcn_wmma_f32_16x16x32_bf16(false, a1, false, qb[1], (short)0, st[t], false, false);
      }
    }

    // ---- additive mask, scaled into base-2 domain ----
    {
      const float* mgk = mg + key0 + hb;
      #pragma unroll
      for (int t = 0; t < 4; ++t)
        #pragma unroll
        for (int r = 0; r < 8; ++r) st[t][r] = fmaf(mgk[t * 16 + r], LOG2E, st[t][r]);
    }
    // ---- last 128x128 block OVERWRITE ----
    if (key0 >= S_LEN - 128 && q0 >= S_LEN - 128) {
      const int kb2 = key0 + hb;
      const int qq  = q0 + ln;
      #pragma unroll
      for (int t = 0; t < 4; ++t)
        #pragma unroll
        for (int r = 0; r < 8; ++r)
          st[t][r] = ((kb2 + t * 16 + r) > qq) ? FMIN : 0.0f;
    }

    // ---- online softmax (base-2): tree max, tree sum, one xor-16 shuffle each ----
    float mz[8];
    #pragma unroll
    for (int r = 0; r < 8; ++r)
      mz[r] = fmaxf(fmaxf(st[0][r], st[1][r]), fmaxf(st[2][r], st[3][r]));
    float mx = fmaxf(fmaxf(fmaxf(mz[0], mz[1]), fmaxf(mz[2], mz[3])),
                     fmaxf(fmaxf(mz[4], mz[5]), fmaxf(mz[6], mz[7])));
    mx = fmaxf(mx, __shfl_xor(mx, 16));
    const float mn    = fmaxf(m_state, mx);
    const float alpha = __builtin_amdgcn_exp2f(m_state - mn);
    m_state = mn;

    float pt[4][8];
    #pragma unroll
    for (int t = 0; t < 4; ++t)
      #pragma unroll
      for (int r = 0; r < 8; ++r) pt[t][r] = __builtin_amdgcn_exp2f(st[t][r] - mn);
    float sz[8];
    #pragma unroll
    for (int r = 0; r < 8; ++r)
      sz[r] = (pt[0][r] + pt[1][r]) + (pt[2][r] + pt[3][r]);
    float rs = (((sz[0] + sz[1]) + (sz[2] + sz[3])) + ((sz[4] + sz[5]) + (sz[6] + sz[7])));
    rs += __shfl_xor(rs, 16);
    l_state = l_state * alpha + rs;
    #pragma unroll
    for (int n = 0; n < 4; ++n)
      #pragma unroll
      for (int r = 0; r < 8; ++r) acc[n][r] *= alpha;

    // ---- P^T: C-layout -> B-layout in registers (two 32-key B tiles) ----
    PB pb[2];
    #pragma unroll
    for (int h2 = 0; h2 < 2; ++h2) {
      unsigned own0[4], own1[4], recv[4];
      #pragma unroll
      for (int j = 0; j < 4; ++j) {
        own0[j] = pack2bf(pt[2*h2][2*j],     pt[2*h2][2*j+1]);
        own1[j] = pack2bf(pt[2*h2+1][2*j],   pt[2*h2+1][2*j+1]);
      }
      #pragma unroll
      for (int j = 0; j < 4; ++j) {
        unsigned send = hi ? own0[j] : own1[j];
        recv[j] = __shfl_xor(send, 16);
      }
      #pragma unroll
      for (int j = 0; j < 4; ++j) {
        pb[h2].u[j]     = hi ? recv[j] : own0[j];
        pb[h2].u[4 + j] = hi ? own1[j] : recv[j];
      }
    }

    // ---- O^T += V^T * P^T : 4 d-tiles x 2 key halves -> 8 WMMA ----
    #pragma unroll
    for (int n = 0; n < 4; ++n) {
      #pragma unroll
      for (int h2 = 0; h2 < 2; ++h2) {
        const int vrow = (n * 16 + ln) * VPITCH + h2 * 32;
        v16bf va = cat16(*(v8bf*)&sV[vrow + 2 * hb],
                         *(v8bf*)&sV[vrow + 2 * hb + 8]);
        acc[n] = __builtin_amdgcn_wmma_f32_16x16x32_bf16(false, va, false, pb[h2].v, (short)0, acc[n], false, false);
      }
    }

    // ---- store next tile into the other buffer, then single barrier ----
    stage_store(nxtb, ns);
    __syncthreads();
  }

  // ---- epilogue: normalize, transpose O^T -> O via wave-private LDS, coalesced stores ----
  {
    float* sO = (float*)smem;              // 8 waves x [16 query rows][OPITCH]
    const int obase = wave * 16 * OPITCH;
    const float invl = 1.0f / l_state;
    #pragma unroll
    for (int n = 0; n < 4; ++n)
      #pragma unroll
      for (int r = 0; r < 8; ++r)
        sO[obase + ln * OPITCH + n * 16 + r + hb] = acc[n][r] * invl;

    asm volatile("s_wait_dscnt 0" ::: "memory");   // wave-private region

    #pragma unroll
    for (int j = 0; j < 8; ++j) {
      const int chunk = j * 32 + lane;
      const int row   = chunk >> 4;
      const int col   = (chunk & 15) * 4;
      float4 val = *(const float4*)&sO[obase + row * OPITCH + col];
      *(float4*)&og[(size_t)(q0 + row) * DHEAD + col] = val;
    }
  }
}

extern "C" void kernel_launch(void* const* d_in, const int* in_sizes, int n_in,
                              void* d_out, int out_size, void* d_ws, size_t ws_size,
                              hipStream_t stream) {
  (void)in_sizes; (void)n_in; (void)out_size; (void)d_ws; (void)ws_size;
  const float* Q  = (const float*)d_in[0];
  const float* K  = (const float*)d_in[1];
  const float* V  = (const float*)d_in[2];
  const float* Mk = (const float*)d_in[3];
  float* O = (float*)d_out;
  fa_bf16_wmma_t_kernel<<<dim3(NBH * NQT), dim3(256), 0, stream>>>(Q, K, V, Mk, O);
}
